// LSTMEncoder_64347200029117
// MI455X (gfx1250) — matrix operand order repaired and checked by reading, not compiled, not run on hardware
//
#include <hip/hip_runtime.h>
#include <hip/hip_bf16.h>

// ---------------- problem constants ----------------
#define VOCAB   32000
#define EMB     256
#define HID     512
#define SEQLEN  2048
#define BATCH   64
#define GATES   (4*HID)          // 2048
#define NWG     16               // cooperating workgroups (one WGP each)
#define WGTHR   256              // 8 wave32s
#define KTOT    (EMB + HID)      // 768 fused K (x-proj + recurrent)
#define KC      (KTOT/32)        // 24 k-chunks of 32
#define NTPW    8                // 8 n-tiles (16 cols) per WG -> 128 cols
#define COLS    128              // gate columns per WG
#define JSL     32               // hidden units per WG (COLS/4)

typedef __attribute__((ext_vector_type(16))) __bf16 v16bf;
typedef __attribute__((ext_vector_type(8)))  float  v8f;

// ---------------- d_ws layout (bytes) ----------------
#define WS_SYNC   0                         // 256 B: [0]=counter [1]=generation
#define WS_HX0    256                       // 64*512 bf16 = 64 KB (h exchange, buf 0)
#define WS_HX1    (256 + 65536)             // 64 KB (buf 1)
#define WS_BIAS   (256 + 2*65536)           // 2048 f32 = 8 KB (b_ih+b_hh)
#define WS_WPACK  (WS_BIAS + 8192)          // 16*8*24*512 bf16 = 3 MB packed weights

// ---------------- LDS layout (bytes) ----------------
#define L_W     0                           // 8*24 packed B tiles * 1KB = 192 KB
#define L_A     196608                      // 4*24 packed A tiles * 1KB = 96 KB
#define L_GATES 196608                      // f32[64][128] = 32 KB, aliases emb part of A
#define L_BIAS  (196608 + 98304)            // f32[128]
#define L_C     (196608 + 98304 + 512)      // f32[64*32] = 8 KB
#define LDS_SZ  (196608 + 98304 + 512 + 8192)   // 303,616 B  (<= 320 KB/WG)

__device__ __forceinline__ unsigned short f2bf(float f) {
    unsigned u = __float_as_uint(f);
    unsigned r = u + 0x7FFFu + ((u >> 16) & 1u);   // round-to-nearest-even
    return (unsigned short)(r >> 16);
}

// K order: A and B both use contiguous k%32 = 16*h + e (no permutation), not the
// hardware's table order; valid because A and B agree. Unlike the A-side
// hardware map (ISA 7.12.2: lanes0-15 hold hw-k {0-7,16-23}, lanes16-31 hold
// {8-15,24-31}), the A placement here is CONTIGUOUS:
//   lane = row + 16*((K%32)>=16), elem = K%16
// The B packing (prep kernel, runs once) uses the same contiguous order.

// ======== prep: pack W_ih|W_hh slices (bf16, B-fragment order) + bias ========
// B packing order (32x16): lane half h, elem e -> k = e + 16*h, col = lane%16.
// Logical K = kc*32 + k is placed there directly (no permutation).
__global__ void lstm_prep_kernel(const float* __restrict__ W_ih,
                                 const float* __restrict__ W_hh,
                                 const float* __restrict__ b_ih,
                                 const float* __restrict__ b_hh,
                                 unsigned short* __restrict__ wpack,
                                 float* __restrict__ bias) {
    int idx = blockIdx.x * blockDim.x + threadIdx.x;
    if (idx < GATES) bias[idx] = b_ih[idx] + b_hh[idx];
    const int total = NWG * NTPW * KC * 512;
    if (idx >= total) return;
    int pos  = idx & 511;
    int lane = pos >> 4, e = pos & 15;
    int kc   = (idx >> 9) % KC;
    int nt   = ((idx >> 9) / KC) % NTPW;
    int wg   = (idx >> 9) / (KC * NTPW);
    int ncol = nt * 16 + (lane & 15);                 // local column 0..127
    int x    = ((lane >> 4) << 4) + e;                // k within the 32-chunk for this slot

    int K    = kc * 32 + x;                           // logical K stored here
    int gate = ncol >> 5;
    int j    = wg * JSL + (ncol & 31);
    int grow = gate * HID + j;                        // row of W (4H-major)
    float w  = (K < EMB) ? W_ih[(size_t)grow * EMB + K]
                         : W_hh[(size_t)grow * HID + (K - EMB)];
    wpack[idx] = f2bf(w);
}

// async copy of 16 bytes global -> LDS (CDNA5 GLOBAL_LOAD_ASYNC_TO_LDS_B128)
__device__ __forceinline__ void async_g2l_b128(void* lds_generic, const void* gaddr) {
    unsigned ldsoff = (unsigned)(unsigned long long)lds_generic;  // low 32 bits = LDS byte addr
    asm volatile("global_load_async_to_lds_b128 %0, %1, off"
                 :: "v"(ldsoff), "v"(gaddr) : "memory");
}
__device__ __forceinline__ void wait_asynccnt0() {
    asm volatile("s_wait_asynccnt 0" ::: "memory");
}

// ======== persistent fused LSTM kernel: 16 WGs, device-scope step barrier ========
__global__ void lstm_scan_kernel(const int*   __restrict__ xs,
                                 const float* __restrict__ emb,
                                 float*       __restrict__ out,
                                 unsigned short* __restrict__ hx0,
                                 unsigned short* __restrict__ hx1,
                                 const float* __restrict__ bias,
                                 const unsigned short* __restrict__ wpack,
                                 unsigned* __restrict__ sync) {
    extern __shared__ char smem[];
    unsigned short* Wlds   = (unsigned short*)(smem + L_W);
    unsigned short* Abuf   = (unsigned short*)(smem + L_A);
    float*          gatesB = (float*)(smem + L_GATES);     // aliases emb region of Abuf
    float*          biasL  = (float*)(smem + L_BIAS);
    float*          cbuf   = (float*)(smem + L_C);

    const int tid  = threadIdx.x;
    const int wg   = blockIdx.x;
    const int wave = tid >> 5;
    const int lane = tid & 31;

    // ---- async-preload this WG's packed weight slice into LDS (loop-invariant) ----
    {
        const unsigned short* src = wpack + (size_t)wg * (NTPW * KC * 512);
        for (int i = tid; i < NTPW * KC * 512 / 8; i += WGTHR)
            async_g2l_b128(Wlds + i * 8, src + i * 8);
    }
    for (int n = tid; n < COLS; n += WGTHR)
        biasL[n] = bias[(n >> 5) * HID + wg * JSL + (n & 31)];
    for (int i = tid; i < BATCH * JSL; i += WGTHR) cbuf[i] = 0.f;  // c0 = 0
    wait_asynccnt0();
    __syncthreads();

    for (int t = 0; t < SEQLEN; ++t) {
        const unsigned short* hxr = (t & 1) ? hx1 : hx0;   // h_t
        unsigned short*       hxw = (t & 1) ? hx0 : hx1;   // h_{t+1}

        // ---- stage emb_t (K 0..255): 8 floats -> 8 bf16 -> one uint4 LDS store ----
        for (int idx = tid; idx < BATCH * EMB / 8; idx += WGTHR) {
            int b  = idx >> 5;                 // 32 8-wide chunks per row
            int k8 = (idx & 31) << 3;          // K start (multiple of 8)
            const float4* src = (const float4*)(emb + (size_t)xs[t * BATCH + b] * EMB + k8);
            float4 f0 = src[0], f1 = src[1];
            uint4 p;
            p.x = (unsigned)f2bf(f0.x) | ((unsigned)f2bf(f0.y) << 16);
            p.y = (unsigned)f2bf(f0.z) | ((unsigned)f2bf(f0.w) << 16);
            p.z = (unsigned)f2bf(f1.x) | ((unsigned)f2bf(f1.y) << 16);
            p.w = (unsigned)f2bf(f1.z) | ((unsigned)f2bf(f1.w) << 16);
            int tile = (k8 >> 5) * 4 + (b >> 4);
            int off  = tile * 512 + ((b & 15) + (((k8 >> 4) & 1) << 4)) * 16 + (k8 & 15);
            *(uint4*)(Abuf + off) = p;
        }
        // ---- stage h_t (K 256..767): async 16-byte global->LDS copies ----
        for (int idx = tid; idx < BATCH * HID / 8; idx += WGTHR) {
            int b  = idx >> 6;                 // 64 8-wide chunks per row
            int j8 = (idx & 63) << 3;          // j start (multiple of 8)
            int tile = (8 + (j8 >> 5)) * 4 + (b >> 4);
            int off  = tile * 512 + ((b & 15) + (((j8 >> 4) & 1) << 4)) * 16 + (j8 & 15);
            if (t == 0) {
                *(uint4*)(Abuf + off) = make_uint4(0u, 0u, 0u, 0u);   // h0 = 0
            } else {
                async_g2l_b128(Abuf + off, hxr + b * HID + j8);
            }
        }
        wait_asynccnt0();
        __syncthreads();

        // ---- GEMM: gates[64 x 128] += A[64 x 768] * Wslice[768 x 128] ----
        v8f acc0 = {}, acc1 = {}, acc2 = {}, acc3 = {};
        const unsigned short* wbase = Wlds + (wave * KC) * 512 + lane * 16;
        const unsigned short* abase = Abuf + lane * 16;
#pragma unroll 4
        for (int kc = 0; kc < KC; ++kc) {
            v16bf bf = *(const v16bf*)(wbase + kc * 512);
            v16bf a0 = *(const v16bf*)(abase + (kc * 4 + 0) * 512);
            v16bf a1 = *(const v16bf*)(abase + (kc * 4 + 1) * 512);
            v16bf a2 = *(const v16bf*)(abase + (kc * 4 + 2) * 512);
            v16bf a3 = *(const v16bf*)(abase + (kc * 4 + 3) * 512);
            acc0 = __builtin_amdgcn_wmma_f32_16x16x32_bf16(false, a0, false, bf, (short)0, acc0, false, false);
            acc1 = __builtin_amdgcn_wmma_f32_16x16x32_bf16(false, a1, false, bf, (short)0, acc1, false, false);
            acc2 = __builtin_amdgcn_wmma_f32_16x16x32_bf16(false, a2, false, bf, (short)0, acc2, false, false);
            acc3 = __builtin_amdgcn_wmma_f32_16x16x32_bf16(false, a3, false, bf, (short)0, acc3, false, false);
        }
        __syncthreads();   // all reads of Abuf(emb) done before gates alias-write

        // ---- scatter D tiles (+bias) to gates LDS: row = Mtile*16 + 8*(lane>=16) + v
        {
            int col   = wave * 16 + (lane & 15);
            int rbase = (lane >> 4) << 3;
            float bc  = biasL[col];
#pragma unroll
            for (int v = 0; v < 8; ++v) {
                gatesB[(0 * 16 + rbase + v) * COLS + col] = acc0[v] + bc;
                gatesB[(1 * 16 + rbase + v) * COLS + col] = acc1[v] + bc;
                gatesB[(2 * 16 + rbase + v) * COLS + col] = acc2[v] + bc;
                gatesB[(3 * 16 + rbase + v) * COLS + col] = acc3[v] + bc;
            }
        }
        __syncthreads();

        // ---- elementwise: i,f,g,o -> c,h ; write hs[t] + h exchange slice ----
        for (int it = tid; it < BATCH * JSL; it += WGTHR) {
            int b = it >> 5, jl = it & 31;
            float gi = gatesB[b * COLS + jl];
            float gf = gatesB[b * COLS + 32 + jl];
            float gg = gatesB[b * COLS + 64 + jl];
            float go = gatesB[b * COLS + 96 + jl];
            float i_ = 1.f / (1.f + __expf(-gi));
            float f_ = 1.f / (1.f + __expf(-gf));
            float g_ = 2.f / (1.f + __expf(-2.f * gg)) - 1.f;   // tanh
            float o_ = 1.f / (1.f + __expf(-go));
            float c  = f_ * cbuf[it] + i_ * g_;
            cbuf[it] = c;
            float th = 2.f / (1.f + __expf(-2.f * c)) - 1.f;
            float h  = o_ * th;
            int j = wg * JSL + jl;
            out[((size_t)t * BATCH + b) * HID + j] = h;
            hxw[b * HID + j] = f2bf(h);
        }

        // ---- device-scope step barrier across the 16 WGs ----
        __syncthreads();
        __builtin_amdgcn_fence(__ATOMIC_RELEASE, "agent");
        if (tid == 0) {
            unsigned gen  = __hip_atomic_load(&sync[1], __ATOMIC_RELAXED, __HIP_MEMORY_SCOPE_AGENT);
            unsigned prev = __hip_atomic_fetch_add(&sync[0], 1u, __ATOMIC_ACQ_REL, __HIP_MEMORY_SCOPE_AGENT);
            if (prev == NWG - 1) {
                __hip_atomic_store(&sync[0], 0u, __ATOMIC_RELAXED, __HIP_MEMORY_SCOPE_AGENT);
                __hip_atomic_store(&sync[1], gen + 1u, __ATOMIC_RELEASE, __HIP_MEMORY_SCOPE_AGENT);
            } else {
                while (__hip_atomic_load(&sync[1], __ATOMIC_RELAXED, __HIP_MEMORY_SCOPE_AGENT) == gen)
                    __builtin_amdgcn_s_sleep(1);
            }
        }
        __syncthreads();
        __builtin_amdgcn_fence(__ATOMIC_ACQUIRE, "agent");  // invalidate WGP$ before reading hx
    }
}

extern "C" void kernel_launch(void* const* d_in, const int* in_sizes, int n_in,
                              void* d_out, int out_size, void* d_ws, size_t ws_size,
                              hipStream_t stream) {
    const int*   xs   = (const int*)  d_in[0];
    const float* emb  = (const float*)d_in[1];
    const float* W_ih = (const float*)d_in[2];
    const float* W_hh = (const float*)d_in[3];
    const float* b_ih = (const float*)d_in[4];
    const float* b_hh = (const float*)d_in[5];
    float* out = (float*)d_out;

    char* ws = (char*)d_ws;
    unsigned*       sync  = (unsigned*)(ws + WS_SYNC);
    unsigned short* hx0   = (unsigned short*)(ws + WS_HX0);
    unsigned short* hx1   = (unsigned short*)(ws + WS_HX1);
    float*          bias  = (float*)(ws + WS_BIAS);
    unsigned short* wpack = (unsigned short*)(ws + WS_WPACK);

    // barrier counter+generation must start at 0 every launch (ws is poisoned once)
    hipMemsetAsync(ws, 0, 256, stream);

    const int total = NWG * NTPW * KC * 512;
    lstm_prep_kernel<<<(total + 255) / 256, 256, 0, stream>>>(W_ih, W_hh, b_ih, b_hh, wpack, bias);

    (void)hipFuncSetAttribute((const void*)lstm_scan_kernel,
                              hipFuncAttributeMaxDynamicSharedMemorySize, (int)LDS_SZ);
    lstm_scan_kernel<<<NWG, WGTHR, LDS_SZ, stream>>>(xs, emb, out, hx0, hx1, bias, wpack, sync);
}
